// SRNO3D_74543452389469
// MI455X (gfx1250) — compile-verified, matrix operand order checked by reading, not run on hardware
//
#include <hip/hip_runtime.h>
#include <math.h>

// ---------------------------------------------------------------------------
// Types / helpers
// ---------------------------------------------------------------------------
typedef __attribute__((ext_vector_type(16))) __bf16          bf16x16;
typedef __attribute__((ext_vector_type(8)))  float           f32x8;
typedef __attribute__((ext_vector_type(8)))  unsigned short  u16x8;
typedef __attribute__((ext_vector_type(16))) unsigned short  u16x16;

#define DEV __device__ __forceinline__

DEV unsigned short f2bf(float f) {
    unsigned int u = __float_as_uint(f);
    u += 0x7FFFu + ((u >> 16) & 1u);           // round-to-nearest-even
    return (unsigned short)(u >> 16);
}
DEV float bf2f(unsigned short h) { return __uint_as_float(((unsigned int)h) << 16); }

DEV u16x8 ld8(const unsigned short* p) { return *(const u16x8*)p; }

DEV bf16x16 mkfrag(u16x8 lo, u16x8 hi) {
    u16x16 c = __builtin_shufflevector(lo, hi, 0,1,2,3,4,5,6,7,8,9,10,11,12,13,14,15);
    return __builtin_bit_cast(bf16x16, c);
}

DEV f32x8 wmma_bf16(bf16x16 a, bf16x16 b, f32x8 c) {
    // D(16x16,f32) = A(16x32,bf16) * B(32x16,bf16) + C
    return __builtin_amdgcn_wmma_f32_16x16x32_bf16(false, a, false, b, (short)0, c, false, false);
}

// async copy of 16 bytes global -> LDS (ASYNCcnt tracked)
DEV void async_g2l_b128(unsigned int ldsByteAddr, const void* gaddr) {
    asm volatile("global_load_async_to_lds_b128 %0, %1, off"
                 :: "v"(ldsByteAddr), "v"((unsigned long long)gaddr) : "memory");
}
DEV void async_wait0() { asm volatile("s_wait_asynccnt 0" ::: "memory"); }
DEV unsigned int lds_addr(const void* p) { return (unsigned int)(unsigned long long)p; }

#define NVOX 32768       // 32^3 encoder voxels
#define NQ   262144      // 64^3 query points
#define PD   34          // padded dim (32 + halo)
#define PP   (PD * PD)   // padded plane
#define PSZ  (PD * PD * PD)

// padded voxel index of interior (z,y,x)
DEV int pidx(int z, int y, int x) { return ((z + 1) * PD + (y + 1)) * PD + (x + 1); }

// ---------------------------------------------------------------------------
// Weight packing
// ---------------------------------------------------------------------------
// conv weights: src f32 [O][32][27] (OIDHW) -> dst bf16 [27][OP][32]
__global__ void k_pack_conv_w(const float* __restrict__ src, unsigned short* __restrict__ dst,
                              int O, int OP) {
    int i = blockIdx.x * 256 + threadIdx.x;
    int total = 27 * OP * 32;
    if (i >= total) return;
    int c = i & 31, o = (i >> 5) % OP, tap = i / (OP * 32);
    float v = (o < O) ? src[(o * 32 + c) * 27 + tap] : 0.f;
    dst[i] = f2bf(v);
}

// dense: src f32 [O][I] -> dst bf16 [OP][KP] (zero padded)
__global__ void k_pack_w(const float* __restrict__ src, unsigned short* __restrict__ dst,
                         int O, int I, int OP, int KP) {
    int i = blockIdx.x * 256 + threadIdx.x;
    if (i >= OP * KP) return;
    int k = i % KP, o = i / KP;
    float v = (o < O && k < I) ? src[o * I + k] : 0.f;
    dst[i] = f2bf(v);
}

__global__ void k_zero16(unsigned short* __restrict__ p, int n) {
    int i = blockIdx.x * 256 + threadIdx.x;
    if (i < n) p[i] = 0;
}

// ---------------------------------------------------------------------------
// EDSR head conv (Cin=1): VALU (only 28 MFLOP); writes padded interior
// ---------------------------------------------------------------------------
__global__ void k_head(const float* __restrict__ in, const float* __restrict__ w,
                       const float* __restrict__ b, unsigned short* __restrict__ out) {
    int i = blockIdx.x * 256 + threadIdx.x;
    if (i >= NVOX * 32) return;
    int oc = i & 31, vox = i >> 5;
    int z = vox >> 10, y = (vox >> 5) & 31, x = vox & 31;
    float acc = b[oc];
#pragma unroll
    for (int t = 0; t < 27; t++) {
        int dz = t / 9 - 1, dy = (t / 3) % 3 - 1, dx = t % 3 - 1;
        int zz = z + dz, yy = y + dy, xx = x + dx;
        if ((unsigned)zz < 32u && (unsigned)yy < 32u && (unsigned)xx < 32u)
            acc += w[oc * 27 + t] * in[(zz << 10) + (yy << 5) + xx];
    }
    out[(size_t)pidx(z, y, x) * 32 + oc] = f2bf(acc);
}

// ---------------------------------------------------------------------------
// 3x3x3 conv, Cin=32, WMMA implicit GEMM over a zero-haloed padded volume.
// act: bf16 [PSZ][32]. wpk: bf16 [27][OP][32] (block tile staged to LDS async).
// B-side: single 64-bit base address per wave; all 27 taps are compile-time
// constant byte deltas -> immediate `offset:` fields on global_load_b128.
// epilogue: v = acc + bias; relu?; v = v*resScale + res?
// ---------------------------------------------------------------------------
__global__ __launch_bounds__(256) void k_conv3(
    const unsigned short* __restrict__ act, const unsigned short* __restrict__ wpk,
    const float* __restrict__ bias, const unsigned short* __restrict__ res,
    unsigned short* __restrict__ out, int OP, int OC, int relu, float resScale,
    int outStride, int outPad) {
    __shared__ unsigned short sCW[27 * 16 * 32];          // 27 KB weight tile
    int ocb = blockIdx.y * 16;
    // ---- async stage this block's 16-row weight tile for all 27 taps ----
    for (int v = threadIdx.x; v < 1728; v += 256) {       // 1728 x 16B = 27648 B
        int tap = v >> 6, rem = v & 63, row = rem >> 2, kc = rem & 3;
        async_g2l_b128(lds_addr(&sCW[(tap * 16 + row) * 32 + kc * 8]),
                       wpk + (size_t)(tap * OP + ocb + row) * 32 + kc * 8);
    }
    async_wait0();
    __syncthreads();

    int wave = threadIdx.x >> 5, lane = threadIdx.x & 31;
    int hv = lane >> 4, col = lane & 15;
    int tile = blockIdx.x * 8 + wave;
    int voxbase = tile * 16;
    int z = voxbase >> 10, y = (voxbase >> 5) & 31, x0 = voxbase & 31;
    int kb0 = hv * 8, kb1 = 16 + hv * 8;
    int myx = x0 + col;
    // one base pointer (center voxel row); taps are constant deltas
    const unsigned short* base = act + ((size_t)pidx(z, y, myx) << 5);
    const unsigned short* wbase = &sCW[col * 32];
    f32x8 acc = {};
#pragma unroll
    for (int dz = -1; dz <= 1; dz++) {
#pragma unroll
        for (int dy = -1; dy <= 1; dy++) {
#pragma unroll
            for (int dx = -1; dx <= 1; dx++) {
                const int tap = (dz + 1) * 9 + (dy + 1) * 3 + (dx + 1);
                const int dlt = (dz * PP + dy * PD + dx) * 32;   // const after unroll
                const unsigned short* wrow = wbase + tap * 16 * 32;
                bf16x16 af = mkfrag(ld8(wrow + kb0), ld8(wrow + kb1));
                const unsigned short* arow = base + dlt;
                bf16x16 bf = mkfrag(ld8(arow + kb0), ld8(arow + kb1));
                acc = wmma_bf16(af, bf, acc);
            }
        }
    }
    int vox = voxbase + col;
    size_t oidx = outPad ? (size_t)pidx(z, y, myx) : (size_t)vox;
    int och0 = ocb + hv * 8;
    u16x8 rv = {0,0,0,0,0,0,0,0};
    if (res && och0 < OC) rv = ld8(res + oidx * outStride + och0);
    u16x8 p;
#pragma unroll
    for (int r = 0; r < 8; r++) {
        int oc = och0 + r;
        float v = acc[r] + ((oc < OC) ? bias[oc] : 0.f);
        if (relu) v = fmaxf(v, 0.f);
        if (res) v = v * resScale + bf2f(rv[r]);
        p[r] = f2bf(v);
    }
    if (och0 < OC) *(u16x8*)(out + oidx * outStride + och0) = p;
}

// ---------------------------------------------------------------------------
// pixel_shuffle3d (faithful to reference transpose) : t bf16 [vox][8] -> feat f32 [64^3]
// ---------------------------------------------------------------------------
__global__ void k_shuffle(const unsigned short* __restrict__ t, float* __restrict__ feat) {
    int f = blockIdx.x * 256 + threadIdx.x;
    if (f >= NQ) return;
    int c  = f >> 15;
    int s1 = (f >> 14) & 1, s2 = (f >> 13) & 1, s3 = (f >> 12) & 1;
    int d2 = (f >> 8) & 15, h2 = (f >> 4) & 15, w2 = f & 15;
    int vox = ((d2 * 2 + s1) << 10) + ((h2 * 2 + s2) << 5) + (w2 * 2 + s3);
    feat[f] = bf2f(t[vox * 8 + c]);
}

// ---------------------------------------------------------------------------
// Build 35-ch grid (padded to 64) : bf16 [N][64]
// ---------------------------------------------------------------------------
__global__ void k_grid(const float* __restrict__ coord, const float* __restrict__ cell,
                       const float* __restrict__ feat, unsigned short* __restrict__ g) {
    int n = blockIdx.x * 256 + threadIdx.x;
    if (n >= NQ) return;
    float cz = coord[n * 3 + 0], cx = coord[n * 3 + 1], cy = coord[n * 3 + 2];
    const float r64 = 1.0f / 64.0f;
    float rel[8][3], fts[8], area[8];
#pragma unroll
    for (int j = 0; j < 8; j++) {
        float vz = (j & 4) ? 1.f : -1.f, vx = (j & 2) ? 1.f : -1.f, vy = (j & 1) ? 1.f : -1.f;
        float az = fminf(fmaxf(cz + vz * r64 + 1e-6f, -1.f + 1e-6f), 1.f - 1e-6f);
        float ax = fminf(fmaxf(cx + vx * r64 + 1e-6f, -1.f + 1e-6f), 1.f - 1e-6f);
        float ay = fminf(fmaxf(cy + vy * r64 + 1e-6f, -1.f + 1e-6f), 1.f - 1e-6f);
        int iz = (int)fminf(fmaxf(rintf(((az + 1.f) * 64.f - 1.f) * 0.5f), 0.f), 63.f);
        int ih = (int)fminf(fmaxf(rintf(((ax + 1.f) * 64.f - 1.f) * 0.5f), 0.f), 63.f);
        int iw = (int)fminf(fmaxf(rintf(((ay + 1.f) * 64.f - 1.f) * 0.5f), 0.f), 63.f);
        fts[j] = feat[(iz << 12) + (ih << 6) + iw];
        float oz = (2.f * iz + 1.f) * r64 - 1.f;
        float ox = (2.f * ih + 1.f) * r64 - 1.f;
        float oy = (2.f * iw + 1.f) * r64 - 1.f;
        rel[j][0] = (cz - oz) * 64.f; rel[j][1] = (cx - ox) * 64.f; rel[j][2] = (cy - oy) * 64.f;
        area[j] = fabsf(rel[j][0] * rel[j][1] * rel[j][2]) + 1e-9f;
    }
    float tot = 0.f;
#pragma unroll
    for (int j = 0; j < 8; j++) tot += area[j];
    const int perm[8] = {2, 3, 4, 1, 0, 5, 6, 7};   // the three sequential swaps
    unsigned short* gr = g + (size_t)n * 64;
#pragma unroll
    for (int j = 0; j < 8; j++) {
        gr[3 * j + 0] = f2bf(rel[j][0]);
        gr[3 * j + 1] = f2bf(rel[j][1]);
        gr[3 * j + 2] = f2bf(rel[j][2]);
    }
#pragma unroll
    for (int j = 0; j < 8; j++) gr[24 + j] = f2bf(fts[j] * (area[perm[j]] / tot));
#pragma unroll
    for (int k = 0; k < 3; k++) gr[32 + k] = f2bf(cell[k] * 64.f);
#pragma unroll
    for (int k = 35; k < 64; k++) gr[k] = 0;
}

// ---------------------------------------------------------------------------
// Generic WMMA GEMM: D[n][oc] = act[n][K] * W[oc][K]^T (+bias, epilogue)
// Weight tile (16 rows x K) staged to LDS via async loads, shared by 8 waves.
// epi: 0=none 1=relu 2=gelu(exact). Optional: residual add (bf16 [n][OCs]),
// qkv routing (out channel 8-block -> q/k/v head-interleaved), f32 scalar out (oc==0).
// Block = 8 waves; wave = 16 points x 16 out-ch; grid = (Npoints/128, OP/16).
// ---------------------------------------------------------------------------
__global__ __launch_bounds__(256) void k_gemm(
    const unsigned short* __restrict__ act, int K,
    const unsigned short* __restrict__ wpk, const float* __restrict__ bias,
    int OC, int OCs,
    unsigned short* __restrict__ out, const unsigned short* __restrict__ res,
    float* __restrict__ f32out,
    unsigned short* __restrict__ qout, unsigned short* __restrict__ kout,
    unsigned short* __restrict__ vout, int epi) {
    __shared__ unsigned short sW[16 * 256];               // up to 8 KB
    int ocb = blockIdx.y * 16;
    int kv8 = K >> 3;
    for (int v = threadIdx.x; v < 16 * kv8; v += 256) {
        int row = v / kv8, kc = v - row * kv8;
        async_g2l_b128(lds_addr(&sW[row * K + kc * 8]),
                       wpk + (size_t)(ocb + row) * K + kc * 8);
    }
    async_wait0();
    __syncthreads();

    int wave = threadIdx.x >> 5, lane = threadIdx.x & 31;
    int hv = lane >> 4, col = lane & 15;
    int n = blockIdx.x * 128 + wave * 16 + col;
    f32x8 acc = {};
    const unsigned short* arow = act + (size_t)n * K;
    const unsigned short* wrow = &sW[col * K];
    for (int k0 = 0; k0 < K; k0 += 32) {
        bf16x16 af = mkfrag(ld8(wrow + k0 + hv * 8), ld8(wrow + k0 + 16 + hv * 8));
        bf16x16 bf = mkfrag(ld8(arow + k0 + hv * 8), ld8(arow + k0 + 16 + hv * 8));
        acc = wmma_bf16(af, bf, acc);
    }
    int och0 = ocb + hv * 8;
    u16x8 rv = {0,0,0,0,0,0,0,0};
    if (res) rv = ld8(res + (size_t)n * OCs + och0);
    float v[8];
#pragma unroll
    for (int r = 0; r < 8; r++) {
        int oc = och0 + r;
        float x = acc[r] + ((oc < OC) ? bias[oc] : 0.f);
        if (epi == 1) x = fmaxf(x, 0.f);
        else if (epi == 2) x = 0.5f * x * (1.f + erff(x * 0.70710678118654752f));
        if (res) x += bf2f(rv[r]);
        v[r] = x;
    }
    if (f32out) { if (och0 == 0) f32out[n] = v[0]; return; }
    u16x8 p;
#pragma unroll
    for (int r = 0; r < 8; r++) p[r] = f2bf(v[r]);
    if (qout) {
        int blk8 = och0 >> 3;         // channel-8 block id within 192
        int sel = blk8 % 3, head = blk8 / 3;
        unsigned short* dst = (sel == 0 ? qout : (sel == 1 ? kout : vout)) + (size_t)n * 64 + head * 8;
        *(u16x8*)dst = p;
    } else if (och0 < OC) {
        *(u16x8*)(out + (size_t)n * OCs + och0) = p;
    }
}

// ---------------------------------------------------------------------------
// LayerNorm (ddof=1, /(std+1e-5)) over hc=8, per (point, head), in-place on k and v
// ---------------------------------------------------------------------------
__global__ void k_ln(unsigned short* __restrict__ kb, unsigned short* __restrict__ vb,
                     const float* __restrict__ kw, const float* __restrict__ kbi,
                     const float* __restrict__ vw, const float* __restrict__ vbi) {
    int i = blockIdx.x * 256 + threadIdx.x;
    if (i >= NQ * 8) return;
    int head = i & 7;
    size_t base = (size_t)(i >> 3) * 64 + head * 8;
    float x[8];
#pragma unroll
    for (int t = 0; t < 2; t++) {
        unsigned short* buf = t ? vb : kb;
        const float* w = t ? vw : kw;
        const float* b = t ? vbi : kbi;
        float m = 0.f;
#pragma unroll
        for (int c = 0; c < 8; c++) { x[c] = bf2f(buf[base + c]); m += x[c]; }
        m *= 0.125f;
        float var = 0.f;
#pragma unroll
        for (int c = 0; c < 8; c++) { float d = x[c] - m; var += d * d; }
        float inv = 1.f / (sqrtf(var / 7.f) + 1e-5f);
#pragma unroll
        for (int c = 0; c < 8; c++)
            buf[base + c] = f2bf(w[head * 8 + c] * ((x[c] - m) * inv) + b[head * 8 + c]);
    }
}

// ---------------------------------------------------------------------------
// kv partials: 256 blocks x 8 waves (wave = head); deterministic two-stage sum
// ---------------------------------------------------------------------------
__global__ __launch_bounds__(256) void k_kv(const unsigned short* __restrict__ kb,
                                            const unsigned short* __restrict__ vb,
                                            float* __restrict__ part) {
    int wave = threadIdx.x >> 5, lane = threadIdx.x & 31;
    int head = wave;
    int base = blockIdx.x * 1024;                       // NQ / 256
    float accv[64];
#pragma unroll
    for (int e = 0; e < 64; e++) accv[e] = 0.f;
    for (int p = lane; p < 1024; p += 32) {
        size_t off = (size_t)(base + p) * 64 + head * 8;
        u16x8 kk8 = ld8(kb + off), vv8 = ld8(vb + off);
        float kk[8], vv[8];
#pragma unroll
        for (int c = 0; c < 8; c++) { kk[c] = bf2f(kk8[c]); vv[c] = bf2f(vv8[c]); }
#pragma unroll
        for (int c = 0; c < 8; c++)
#pragma unroll
            for (int d = 0; d < 8; d++) accv[c * 8 + d] += kk[c] * vv[d];
    }
#pragma unroll
    for (int e = 0; e < 64; e++) {
        float s = accv[e];
        for (int o = 16; o > 0; o >>= 1) s += __shfl_xor(s, o, 32);
        if (lane == 0) part[(size_t)blockIdx.x * 512 + head * 64 + e] = s;
    }
}

__global__ void k_kvsum(const float* __restrict__ part, float* __restrict__ kv) {
    int e = blockIdx.x * 256 + threadIdx.x;
    if (e >= 512) return;
    float s = 0.f;
    for (int b = 0; b < 256; b++) s += part[(size_t)b * 512 + e];   // fixed order
    kv[e] = s * (1.0f / (float)NQ);
}

// ---------------------------------------------------------------------------
// o = q * kv ; ret = o + bias(X)
// ---------------------------------------------------------------------------
__global__ __launch_bounds__(256) void k_attno(const unsigned short* __restrict__ qb,
                                               const float* __restrict__ kv,
                                               const unsigned short* __restrict__ xb,
                                               unsigned short* __restrict__ ret) {
    __shared__ float kvs[512];
    for (int i = threadIdx.x; i < 512; i += 256) kvs[i] = kv[i];
    __syncthreads();
    int n = blockIdx.x * 256 + threadIdx.x;
    if (n >= NQ) return;
    size_t b = (size_t)n * 64;
    for (int head = 0; head < 8; head++) {
        u16x8 q8 = ld8(qb + b + head * 8);
        u16x8 x8 = ld8(xb + b + head * 8);
        float q[8];
#pragma unroll
        for (int c = 0; c < 8; c++) q[c] = bf2f(q8[c]);
        u16x8 p;
#pragma unroll
        for (int d = 0; d < 8; d++) {
            float o = 0.f;
#pragma unroll
            for (int c = 0; c < 8; c++) o += q[c] * kvs[head * 64 + c * 8 + d];
            p[d] = f2bf(o + bf2f(x8[d]));
        }
        *(u16x8*)(ret + b + head * 8) = p;
    }
}

// ---------------------------------------------------------------------------
// final: out[n] = fc2[n] + trilinear_border(inp, coord[n])
// ---------------------------------------------------------------------------
DEV void triprep(float c, int s, int& i0, int& i1, float& t) {
    float ix = fminf(fmaxf(((c + 1.f) * (float)s - 1.f) * 0.5f, 0.f), (float)(s - 1));
    float f0 = floorf(ix);
    t = ix - f0;
    i0 = (int)f0;
    i1 = (i0 + 1 < s - 1) ? i0 + 1 : s - 1;
}

__global__ void k_final(const float* __restrict__ fc2v, const float* __restrict__ inp,
                        const float* __restrict__ coord, float* __restrict__ out) {
    int n = blockIdx.x * 256 + threadIdx.x;
    if (n >= NQ) return;
    int z0, z1, x0, x1, y0, y1;
    float tz, tx, ty;
    triprep(coord[n * 3 + 0], 32, z0, z1, tz);
    triprep(coord[n * 3 + 1], 32, x0, x1, tx);
    triprep(coord[n * 3 + 2], 32, y0, y1, ty);
    float acc = 0.f;
#pragma unroll
    for (int a = 0; a < 2; a++) {
        int iz = a ? z1 : z0; float wz = a ? tz : 1.f - tz;
#pragma unroll
        for (int b = 0; b < 2; b++) {
            int ih = b ? x1 : x0; float wh = b ? tx : 1.f - tx;
#pragma unroll
            for (int c = 0; c < 2; c++) {
                int iw = c ? y1 : y0; float ww = c ? ty : 1.f - ty;
                acc += inp[(iz << 10) + (ih << 5) + iw] * (wz * wh * ww);
            }
        }
    }
    out[n] = fc2v[n] + acc;
}

// ---------------------------------------------------------------------------
// Host: orchestrate the whole pipeline on `stream`
// ---------------------------------------------------------------------------
extern "C" void kernel_launch(void* const* d_in, const int* in_sizes, int n_in,
                              void* d_out, int out_size, void* d_ws, size_t ws_size,
                              hipStream_t stream) {
    (void)in_sizes; (void)n_in; (void)out_size; (void)ws_size;
    const float* inp     = (const float*)d_in[0];
    const float* coord   = (const float*)d_in[1];
    const float* cell    = (const float*)d_in[2];
    const float* head_w  = (const float*)d_in[3];
    const float* head_b  = (const float*)d_in[4];
    const float* cw[9];  const float* cb[9];
    for (int b = 0; b < 4; b++) {
        cw[b * 2 + 0] = (const float*)d_in[5 + 4 * b + 0]; cb[b * 2 + 0] = (const float*)d_in[5 + 4 * b + 1];
        cw[b * 2 + 1] = (const float*)d_in[5 + 4 * b + 2]; cb[b * 2 + 1] = (const float*)d_in[5 + 4 * b + 3];
    }
    cw[8] = (const float*)d_in[21]; cb[8] = (const float*)d_in[22];
    const float* tail_w  = (const float*)d_in[23];
    const float* tail_b  = (const float*)d_in[24];
    const float* c00_w   = (const float*)d_in[25];
    const float* c00_b   = (const float*)d_in[26];
    const float *qkv_w[2], *qkv_b[2], *kln_w[2], *kln_b[2], *vln_w[2], *vln_b[2],
                *o1_w[2], *o1_b[2], *o2_w[2], *o2_b[2];
    for (int i = 0; i < 2; i++) {
        int base = 27 + 10 * i;
        qkv_w[i] = (const float*)d_in[base + 0]; qkv_b[i] = (const float*)d_in[base + 1];
        kln_w[i] = (const float*)d_in[base + 2]; kln_b[i] = (const float*)d_in[base + 3];
        vln_w[i] = (const float*)d_in[base + 4]; vln_b[i] = (const float*)d_in[base + 5];
        o1_w[i]  = (const float*)d_in[base + 6]; o1_b[i]  = (const float*)d_in[base + 7];
        o2_w[i]  = (const float*)d_in[base + 8]; o2_b[i]  = (const float*)d_in[base + 9];
    }
    const float* fc1_w = (const float*)d_in[47];
    const float* fc1_b = (const float*)d_in[48];
    const float* fc2_w = (const float*)d_in[49];
    const float* fc2_b = (const float*)d_in[50];
    float* out = (float*)d_out;

    // ---- workspace carving ----
    char* wsp = (char*)d_ws; size_t off = 0;
    auto carve = [&](size_t bytes) -> void* {
        void* p = wsp + off; off += (bytes + 255) & ~(size_t)255; return p;
    };
    unsigned short* wpC[9];
    for (int i = 0; i < 9; i++) wpC[i] = (unsigned short*)carve(27 * 32 * 32 * 2);
    unsigned short* wpT   = (unsigned short*)carve(27 * 16 * 32 * 2);
    unsigned short* wc00  = (unsigned short*)carve(64 * 64 * 2);
    unsigned short* wqkv[2], *wo1[2], *wo2[2];
    for (int i = 0; i < 2; i++) {
        wqkv[i] = (unsigned short*)carve(192 * 64 * 2);
        wo1[i]  = (unsigned short*)carve(64 * 64 * 2);
        wo2[i]  = (unsigned short*)carve(64 * 64 * 2);
    }
    unsigned short* wfc1 = (unsigned short*)carve(256 * 64 * 2);
    unsigned short* wfc2 = (unsigned short*)carve(16 * 256 * 2);
    const size_t EPAD = (size_t)PSZ * 32;                                // padded elems
    unsigned short* E0 = (unsigned short*)carve(EPAD * 2);               // x (head out)
    unsigned short* E1 = (unsigned short*)carve(EPAD * 2);               // r
    unsigned short* E2 = (unsigned short*)carve(EPAD * 2);               // tmp
    unsigned short* Tl = (unsigned short*)carve((size_t)NVOX * 8 * 2);   // tail out
    float*          ft = (float*)carve((size_t)NQ * 4);                  // feature f32
    unsigned short* B0 = (unsigned short*)carve((size_t)NQ * 64 * 2);    // grid -> h
    unsigned short* B1 = (unsigned short*)carve((size_t)NQ * 64 * 2);    // X (in-place attn)
    unsigned short* Bq = (unsigned short*)carve((size_t)NQ * 64 * 2);
    unsigned short* Bk = (unsigned short*)carve((size_t)NQ * 64 * 2);
    unsigned short* Bv = (unsigned short*)carve((size_t)NQ * 64 * 2);
    unsigned short* Br = (unsigned short*)carve((size_t)NQ * 64 * 2);    // ret
    unsigned short* FC = (unsigned short*)carve((size_t)32768 * 256 * 2);
    float*          f2 = (float*)carve((size_t)NQ * 4);
    float*          kvp = (float*)carve((size_t)256 * 512 * 4);
    float*          kvb = (float*)carve(512 * 4);

    // ---- pack weights ----
    for (int i = 0; i < 9; i++)
        hipLaunchKernelGGL(k_pack_conv_w, dim3(108), dim3(256), 0, stream, cw[i], wpC[i], 32, 32);
    hipLaunchKernelGGL(k_pack_conv_w, dim3(54), dim3(256), 0, stream, tail_w, wpT, 8, 16);
    hipLaunchKernelGGL(k_pack_w, dim3(16), dim3(256), 0, stream, c00_w, wc00, 64, 35, 64, 64);
    for (int i = 0; i < 2; i++) {
        hipLaunchKernelGGL(k_pack_w, dim3(48), dim3(256), 0, stream, qkv_w[i], wqkv[i], 192, 64, 192, 64);
        hipLaunchKernelGGL(k_pack_w, dim3(16), dim3(256), 0, stream, o1_w[i], wo1[i], 64, 64, 64, 64);
        hipLaunchKernelGGL(k_pack_w, dim3(16), dim3(256), 0, stream, o2_w[i], wo2[i], 64, 64, 64, 64);
    }
    hipLaunchKernelGGL(k_pack_w, dim3(64), dim3(256), 0, stream, fc1_w, wfc1, 256, 64, 256, 64);
    hipLaunchKernelGGL(k_pack_w, dim3(16), dim3(256), 0, stream, fc2_w, wfc2, 1, 256, 16, 256);

    // ---- EDSR ----
    int zb = (int)((EPAD + 255) / 256);
    hipLaunchKernelGGL(k_zero16, dim3(zb), dim3(256), 0, stream, E0, (int)EPAD); // halo zeros
    hipLaunchKernelGGL(k_zero16, dim3(zb), dim3(256), 0, stream, E2, (int)EPAD);
    hipLaunchKernelGGL(k_head, dim3(4096), dim3(256), 0, stream, inp, head_w, head_b, E0);
    hipMemcpyAsync(E1, E0, EPAD * 2, hipMemcpyDeviceToDevice, stream);   // r = x (halo too)
    for (int b = 0; b < 4; b++) {
        hipLaunchKernelGGL(k_conv3, dim3(256, 2), dim3(256), 0, stream,
                           E1, wpC[b * 2 + 0], cb[b * 2 + 0], (const unsigned short*)nullptr,
                           E2, 32, 32, 1, 1.f, 32, 1);
        hipLaunchKernelGGL(k_conv3, dim3(256, 2), dim3(256), 0, stream,
                           E2, wpC[b * 2 + 1], cb[b * 2 + 1], E1, E1, 32, 32, 1, 2.f, 32, 1);
    }
    hipLaunchKernelGGL(k_conv3, dim3(256, 2), dim3(256), 0, stream,
                       E1, wpC[8], cb[8], E0, E2, 32, 32, 0, 1.f, 32, 1);
    hipLaunchKernelGGL(k_conv3, dim3(256, 1), dim3(256), 0, stream,
                       E2, wpT, tail_b, (const unsigned short*)nullptr, Tl, 16, 8, 0, 1.f, 8, 0);
    hipLaunchKernelGGL(k_shuffle, dim3(1024), dim3(256), 0, stream, Tl, ft);

    // ---- decoder ----
    hipLaunchKernelGGL(k_grid, dim3(1024), dim3(256), 0, stream, coord, cell, ft, B0);
    hipLaunchKernelGGL(k_gemm, dim3(2048, 4), dim3(256), 0, stream,
                       B0, 64, wc00, c00_b, 64, 64, B1, (const unsigned short*)nullptr,
                       (float*)nullptr, (unsigned short*)nullptr, (unsigned short*)nullptr,
                       (unsigned short*)nullptr, 0);
    for (int i = 0; i < 2; i++) {
        hipLaunchKernelGGL(k_gemm, dim3(2048, 12), dim3(256), 0, stream,
                           B1, 64, wqkv[i], qkv_b[i], 192, 64, (unsigned short*)nullptr,
                           (const unsigned short*)nullptr, (float*)nullptr, Bq, Bk, Bv, 0);
        hipLaunchKernelGGL(k_ln, dim3(8192), dim3(256), 0, stream,
                           Bk, Bv, kln_w[i], kln_b[i], vln_w[i], vln_b[i]);
        hipLaunchKernelGGL(k_kv, dim3(256), dim3(256), 0, stream, Bk, Bv, kvp);
        hipLaunchKernelGGL(k_kvsum, dim3(2), dim3(256), 0, stream, kvp, kvb);
        hipLaunchKernelGGL(k_attno, dim3(1024), dim3(256), 0, stream, Bq, kvb, B1, Br);
        hipLaunchKernelGGL(k_gemm, dim3(2048, 4), dim3(256), 0, stream,
                           Br, 64, wo1[i], o1_b[i], 64, 64, B0, (const unsigned short*)nullptr,
                           (float*)nullptr, (unsigned short*)nullptr, (unsigned short*)nullptr,
                           (unsigned short*)nullptr, 2);
        hipLaunchKernelGGL(k_gemm, dim3(2048, 4), dim3(256), 0, stream,
                           B0, 64, wo2[i], o2_b[i], 64, 64, B1, B1,
                           (float*)nullptr, (unsigned short*)nullptr, (unsigned short*)nullptr,
                           (unsigned short*)nullptr, 0);
    }
    for (int c0 = 0; c0 < NQ; c0 += 32768) {
        hipLaunchKernelGGL(k_gemm, dim3(256, 16), dim3(256), 0, stream,
                           B1 + (size_t)c0 * 64, 64, wfc1, fc1_b, 256, 256, FC,
                           (const unsigned short*)nullptr, (float*)nullptr,
                           (unsigned short*)nullptr, (unsigned short*)nullptr,
                           (unsigned short*)nullptr, 2);
        hipLaunchKernelGGL(k_gemm, dim3(256, 1), dim3(256), 0, stream,
                           FC, 256, wfc2, fc2_b, 1, 16, (unsigned short*)nullptr,
                           (const unsigned short*)nullptr, f2 + c0,
                           (unsigned short*)nullptr, (unsigned short*)nullptr,
                           (unsigned short*)nullptr, 0);
    }
    hipLaunchKernelGGL(k_final, dim3(1024), dim3(256), 0, stream, f2, inp, coord, out);
}
